// DeformConvBlock_34333968564804
// MI455X (gfx1250) — matrix operand order repaired and checked by reading, not compiled, not run on hardware
//
#include <hip/hip_runtime.h>
#include <hip/hip_bf16.h>

typedef __bf16 bf16_t;
typedef bf16_t v16bf __attribute__((ext_vector_type(16)));
typedef float  v8f   __attribute__((ext_vector_type(8)));

__device__ __forceinline__ bf16_t f32_to_bf16(float f) {
    unsigned u = __builtin_bit_cast(unsigned, f);
    u += 0x7FFFu + ((u >> 16) & 1u);           // round-to-nearest-even
    unsigned short h = (unsigned short)(u >> 16);
    return __builtin_bit_cast(bf16_t, h);
}

// CDNA5 async global->LDS copy (ASYNCcnt path, ISA 08_async_tensor §4).
__device__ __forceinline__ void async_copy_b128(unsigned lds_off, const bf16_t* gaddr) {
    asm volatile("global_load_async_to_lds_b128 %0, %1, off"
                 :: "v"(lds_off), "v"((unsigned long long)(size_t)gaddr)
                 : "memory");
}
__device__ __forceinline__ void wait_async_le0() {
    asm volatile("s_wait_asynccnt 0x0" ::: "memory");
}
__device__ __forceinline__ void wait_async_le1() {
    asm volatile("s_wait_asynccnt 0x1" ::: "memory");   // allow 1 in flight
}

// ---------------------------------------------------------------------------
// Swizzled fragment addressing: matrices stored directly in WMMA fragment
// order so each lane loads one contiguous v16bf (32B).
//   A (16x32 bf16, ISA 7.12.2): lane = (m&15) + 16*((ki>>3)&1),
//                               j    = (ki&7) + 8*(ki>>4)
//   B (32x16 bf16):             lane = (n&15) + 16*((ki>>3)&1),
//                               j    = (ki&7) + 8*(ki>>4)
// Tiles stored as [tile][lane][j], 512 elements (1KB) per tile.
// ---------------------------------------------------------------------------
__device__ __forceinline__ size_t a_swizzle(int m, int k, int ktiles) {
    int mi = m & 15, ki = k & 31;
    int lane = mi + (((ki >> 3) & 1) << 4);
    int j = (ki & 7) + ((ki >> 4) << 3);
    return (((size_t)(m >> 4) * ktiles + (k >> 5)) << 9) + (lane << 4) + j;
}
__device__ __forceinline__ size_t b_swizzle(int k, int n, int ntiles) {
    int ki = k & 31, ni = n & 15;
    int lane = ni + (((ki >> 3) & 1) << 4);
    int j = (ki & 7) + ((ki >> 4) << 3);
    return (((size_t)(k >> 5) * ntiles + (n >> 4)) << 9) + (lane << 4) + j;
}

// ---------------------------------------------------------------------------
// Standard 3x3 im2col (pad=1) -> swizzled bf16 A (M x K), k = n*C + c.
// ---------------------------------------------------------------------------
__global__ void k_im2col_std(const float* __restrict__ x, bf16_t* __restrict__ A,
                             int B, int C, int H, int W, int stride, int ho, int wo)
{
    int idx = blockIdx.x * blockDim.x + threadIdx.x;
    int total = B * ho * wo * 9;
    if (idx >= total) return;
    int n = idx % 9; int t = idx / 9;
    int ow = t % wo; t /= wo;
    int oh = t % ho; int b = t / ho;

    int ih = oh * stride - 1 + n / 3;
    int iw = ow * stride - 1 + n % 3;
    bool inb = ((unsigned)ih < (unsigned)H) && ((unsigned)iw < (unsigned)W);
    const float* xb = x + (size_t)b * C * H * W + (inb ? (ih * W + iw) : 0);
    int mrow = (b * ho + oh) * wo + ow;
    int K = C * 9, ktiles = K >> 5;
    for (int c = 0; c < C; ++c) {
        float v = inb ? xb[(size_t)c * H * W] : 0.0f;
        A[a_swizzle(mrow, n * C + c, ktiles)] = f32_to_bf16(v);
    }
}

// ---------------------------------------------------------------------------
// DCNv2 bilinear gather + modulation -> swizzled bf16 A (M x K), k = n*C + c.
// ---------------------------------------------------------------------------
__global__ void k_im2col_dcn(const float* __restrict__ x,
                             const float* __restrict__ off, const float* __restrict__ mod,
                             bf16_t* __restrict__ A,
                             int B, int C, int H, int W, int stride, int ho, int wo)
{
    int idx = blockIdx.x * blockDim.x + threadIdx.x;
    int total = B * ho * wo * 9;
    if (idx >= total) return;
    int n = idx % 9; int t = idx / 9;
    int ow = t % wo; t /= wo;
    int oh = t % ho; int b = t / ho;

    size_t plane = (size_t)ho * wo;
    size_t pix = (size_t)oh * wo + ow;
    float oy = off[((size_t)b * 18 + n) * plane + pix];
    float ox = off[((size_t)b * 18 + 9 + n) * plane + pix];
    float mm = mod[((size_t)b * 9 + n) * plane + pix];

    int dy = n / 3 - 1, dx = n % 3 - 1;
    int Hp = H + 2, Wp = W + 2;
    float py = oy + (float)dy + 1.0f + (float)(stride * oh);
    float px = ox + (float)dx + 1.0f + (float)(stride * ow);

    float fpy = floorf(py), fpx = floorf(px);
    int qy0 = min(max((int)fpy, 0), Hp - 1);
    int qx0 = min(max((int)fpx, 0), Wp - 1);
    int qy1 = min(max((int)fpy + 1, 0), Hp - 1);
    int qx1 = min(max((int)fpx + 1, 0), Wp - 1);
    py = fminf(fmaxf(py, 0.0f), (float)(Hp - 1));
    px = fminf(fmaxf(px, 0.0f), (float)(Wp - 1));
    float fy0 = (float)qy0, fx0 = (float)qx0, fy1 = (float)qy1, fx1 = (float)qx1;
    float g_lt = (1.0f + (fy0 - py)) * (1.0f + (fx0 - px));
    float g_rb = (1.0f - (fy1 - py)) * (1.0f - (fx1 - px));
    float g_lb = (1.0f + (fy0 - py)) * (1.0f - (fx1 - px));
    float g_rt = (1.0f - (fy1 - py)) * (1.0f + (fx0 - px));

    int sy0 = qy0 - 1, sx0 = qx0 - 1, sy1 = qy1 - 1, sx1 = qx1 - 1;
    bool vy0 = (unsigned)sy0 < (unsigned)H, vx0 = (unsigned)sx0 < (unsigned)W;
    bool vy1 = (unsigned)sy1 < (unsigned)H, vx1 = (unsigned)sx1 < (unsigned)W;

    const float* xb = x + (size_t)b * C * H * W;
    int mrow = (b * ho + oh) * wo + ow;
    int K = C * 9, ktiles = K >> 5;

    for (int c = 0; c < C; ++c) {
        const float* xc = xb + (size_t)c * H * W;
        float s00 = (vy0 && vx0) ? xc[sy0 * W + sx0] : 0.0f;
        float s11 = (vy1 && vx1) ? xc[sy1 * W + sx1] : 0.0f;
        float s01 = (vy0 && vx1) ? xc[sy0 * W + sx1] : 0.0f;
        float s10 = (vy1 && vx0) ? xc[sy1 * W + sx0] : 0.0f;
        float v = (g_lt * s00 + g_rb * s11 + g_lb * s01 + g_rt * s10) * mm;
        A[a_swizzle(mrow, n * C + c, ktiles)] = f32_to_bf16(v);
    }
}

// ---------------------------------------------------------------------------
// Weight packing (f32 -> swizzled bf16 B, K x N, k = n*C + c).
// ---------------------------------------------------------------------------
__global__ void k_packw(const float* __restrict__ w, bf16_t* __restrict__ Bm,
                        int C, int O)
{
    int idx = blockIdx.x * blockDim.x + threadIdx.x;
    if (idx >= O * C * 9) return;
    int n = idx % 9; int t = idx / 9;
    int c = t % C;   int o = t / C;
    Bm[b_swizzle(n * C + c, o, O >> 4)] = f32_to_bf16(w[((size_t)o * C + c) * 9 + n]);
}

// Pack offset (18ch) + mask (9ch) weights into one N=32 (padded) B matrix.
__global__ void k_packw_offmask(const float* __restrict__ wp, const float* __restrict__ wm,
                                bf16_t* __restrict__ Bm, int C)
{
    int idx = blockIdx.x * blockDim.x + threadIdx.x;
    if (idx >= 32 * C * 9) return;
    int n = idx % 9; int t = idx / 9;
    int c = t % C;   int o = t / C;
    float v = 0.0f;
    if (o < 18)      v = wp[((size_t)o * C + c) * 9 + n];
    else if (o < 27) v = wm[((size_t)(o - 18) * C + c) * 9 + n];
    Bm[b_swizzle(n * C + c, o, 2)] = f32_to_bf16(v);
}

// ---------------------------------------------------------------------------
// Issue async stage of 4 A tiles (4 KB) for K-step kt into buffer `buf`.
// 256 threads x 16B each.
// ---------------------------------------------------------------------------
__device__ __forceinline__ void issue_stage(const bf16_t* A, unsigned ldsbase,
                                            size_t mt4, int kt, int ktiles,
                                            int tid, int buf)
{
    int sub = tid >> 6;            // which of the 4 M-subtiles
    int within = tid & 63;         // 64 threads x 16B = 1KB tile
    const bf16_t* src = A + (((mt4 + sub) * (size_t)ktiles + kt) << 9) + within * 8;
    async_copy_b128(ldsbase + (unsigned)(buf * 4096 + sub * 1024 + within * 16), src);
}

// ---------------------------------------------------------------------------
// Main WMMA GEMM (O=128): BM=64 per block, 8 waves = 8 N-tiles, 4 WMMA/K-step
// per wave. Double-buffered async global->LDS staging hides the copy behind
// the WMMAs. Fused conv-bias + BN (+residual, ReLU) epilogue.
// ---------------------------------------------------------------------------
__global__ void __launch_bounds__(256)
k_gemm_dcn(const bf16_t* __restrict__ A, const bf16_t* __restrict__ Bm,
           const float* __restrict__ convb,
           const float* __restrict__ bng, const float* __restrict__ bnb,
           const float* __restrict__ bnm, const float* __restrict__ bnv,
           const float* __restrict__ addsrc, float* __restrict__ out,
           int M, int K, int O, int ho, int wo, int relu)
{
    int tid = threadIdx.x;
    int lane = tid & 31;
    int wave = tid >> 5;                 // N tile index (O=128 -> 8 tiles)
    int ktiles = K >> 5;
    int ntiles = O >> 4;
    size_t mt4 = (size_t)blockIdx.x * 4;

    __shared__ __align__(32) bf16_t As[2 * 4 * 512];     // double buffer, 8KB
    unsigned ldsbase = (unsigned)(size_t)(void*)As;

    v8f acc[4] = {};
    issue_stage(A, ldsbase, mt4, 0, ktiles, tid, 0);
    for (int kt = 0; kt < ktiles; ++kt) {
        int cur = kt & 1;
        if (kt + 1 < ktiles) {
            issue_stage(A, ldsbase, mt4, kt + 1, ktiles, tid, cur ^ 1);
            __builtin_prefetch(Bm + (((size_t)(kt + 1) * ntiles + wave) << 9), 0, 3);
            wait_async_le1();            // current buffer landed, next in flight
        } else {
            wait_async_le0();
        }
        __syncthreads();

        const bf16_t* Asc = As + cur * 2048;
        v16bf b = *(reinterpret_cast<const v16bf*>(
                        Bm + (((size_t)kt * ntiles + wave) << 9)) + lane);
#pragma unroll
        for (int s = 0; s < 4; ++s) {
            v16bf a = *(reinterpret_cast<const v16bf*>(Asc + s * 512) + lane);
            acc[s] = __builtin_amdgcn_wmma_f32_16x16x32_bf16(
                         false, a, false, b, (short)0, acc[s], false, false);
        }
        __syncthreads();
    }

    // Epilogue. D fragment: M = r + 8*(lane>=16), N = lane&15.
    int o = (wave << 4) + (lane & 15);
    float scale = bng[o] * rsqrtf(bnv[o] + 1e-5f);
    float shift = bnb[o] - bnm[o] * scale + convb[o] * scale;
    int mhi = (lane >> 4) << 3;
#pragma unroll
    for (int s = 0; s < 4; ++s) {
#pragma unroll
        for (int r = 0; r < 8; ++r) {
            int m = (int)((mt4 + s) << 4) + r + mhi;
            if (m >= M) continue;
            float v = acc[s][r] * scale + shift;
            int ow = m % wo; int t2 = m / wo;
            int oh = t2 % ho; int b = t2 / ho;
            size_t oidx = (((size_t)b * O + o) * ho + oh) * wo + ow;
            if (addsrc) v += addsrc[oidx];
            if (relu) v = fmaxf(v, 0.0f);
            out[oidx] = v;
        }
    }
}

// ---------------------------------------------------------------------------
// Offset+mask WMMA GEMM (N padded to 32): wave w -> (msub = w>>1, nt = w&1).
// Same double-buffered async staging. Epilogue: ch 0-17 -> offsets,
// 18-26 -> sigmoid mask, 27-31 discarded.
// ---------------------------------------------------------------------------
__global__ void __launch_bounds__(256)
k_gemm_offmask(const bf16_t* __restrict__ A, const bf16_t* __restrict__ Bm,
               const float* __restrict__ bp, const float* __restrict__ bmask,
               float* __restrict__ offb, float* __restrict__ modb,
               int M, int K, int ho, int wo)
{
    int tid = threadIdx.x;
    int lane = tid & 31;
    int wave = tid >> 5;
    int msub = wave >> 1, nt = wave & 1;
    int ktiles = K >> 5;
    size_t mt4 = (size_t)blockIdx.x * 4;

    __shared__ __align__(32) bf16_t As[2 * 4 * 512];
    unsigned ldsbase = (unsigned)(size_t)(void*)As;

    v8f acc = {};
    issue_stage(A, ldsbase, mt4, 0, ktiles, tid, 0);
    for (int kt = 0; kt < ktiles; ++kt) {
        int cur = kt & 1;
        if (kt + 1 < ktiles) {
            issue_stage(A, ldsbase, mt4, kt + 1, ktiles, tid, cur ^ 1);
            __builtin_prefetch(Bm + (((size_t)(kt + 1) * 2 + nt) << 9), 0, 3);
            wait_async_le1();
        } else {
            wait_async_le0();
        }
        __syncthreads();

        v16bf b = *(reinterpret_cast<const v16bf*>(
                        Bm + (((size_t)kt * 2 + nt) << 9)) + lane);
        v16bf a = *(reinterpret_cast<const v16bf*>(
                        As + cur * 2048 + msub * 512) + lane);
        acc = __builtin_amdgcn_wmma_f32_16x16x32_bf16(
                  false, a, false, b, (short)0, acc, false, false);
        __syncthreads();
    }

    int o = (nt << 4) + (lane & 15);
    if (o < 27) {
        float bias = (o < 18) ? bp[o] : bmask[o - 18];
        int mhi = (lane >> 4) << 3;
#pragma unroll
        for (int r = 0; r < 8; ++r) {
            int m = (int)((mt4 + msub) << 4) + r + mhi;
            if (m >= M) continue;
            float v = acc[r] + bias;
            int ow = m % wo; int t2 = m / wo;
            int oh = t2 % ho; int b = t2 / ho;
            if (o < 18)
                offb[(((size_t)b * 18 + o) * ho + oh) * wo + ow] = v;
            else
                modb[(((size_t)b * 9 + (o - 18)) * ho + oh) * wo + ow] =
                    1.0f / (1.0f + __expf(-v));
        }
    }
}

// ---------------------------------------------------------------------------
// Host orchestration.
// Input order: x, dc1{w,b,wp,bp,wm,bm}, dc2, dc3, dcid, bn1{g,b,m,v}, bn2,
// bn3, bnid.
// ---------------------------------------------------------------------------
extern "C" void kernel_launch(void* const* d_in, const int* in_sizes, int n_in,
                              void* d_out, int out_size, void* d_ws, size_t ws_size,
                              hipStream_t stream) {
    (void)in_sizes; (void)n_in; (void)out_size; (void)ws_size;
    const int B = 4, C = 128, O = 128;

    const float* x = (const float*)d_in[0];
    const float* dc[4][6];   // w, b, wp, bp, wm, bm
    for (int i = 0; i < 4; ++i)
        for (int j = 0; j < 6; ++j)
            dc[i][j] = (const float*)d_in[1 + i * 6 + j];
    const float* bn[4][4];   // g, b, m, v
    for (int i = 0; i < 4; ++i)
        for (int j = 0; j < 4; ++j)
            bn[i][j] = (const float*)d_in[25 + i * 4 + j];

    // Workspace layout (all offsets 32B-aligned).
    float*  y1    = (float*)d_ws;                        // 4*128*64*64
    float*  y2    = y1 + (size_t)4 * 128 * 64 * 64;      // 4*128*32*32
    float*  y3    = y2 + (size_t)4 * 128 * 32 * 32;
    float*  offb  = y3 + (size_t)4 * 128 * 32 * 32;      // up to 4*18*64*64
    float*  modb  = offb + (size_t)4 * 18 * 64 * 64;     // up to 4*9*64*64
    bf16_t* Abuf  = (bf16_t*)(modb + (size_t)4 * 9 * 64 * 64); // 16384*1152 (shared by both GEMM stages)
    bf16_t* Bbuf  = Abuf + (size_t)16384 * 1152;               // 1152*128
    bf16_t* Bbuf2 = Bbuf + (size_t)1152 * 128;                 // 1152*32

    auto run_dconv = [&](const float* xin, int H, int W, int stride,
                         const float* const* p, const float* const* q,
                         const float* addsrc, float* yout, int relu) {
        int ho = (H + 2 - 3) / stride + 1;
        int wo = (W + 2 - 3) / stride + 1;
        int M = B * ho * wo, K = C * 9;
        int tIm = B * ho * wo * 9;

        // Stage 1: offsets + mask via WMMA GEMM (N padded to 32).
        k_im2col_std<<<(tIm + 255) / 256, 256, 0, stream>>>(
            xin, Abuf, B, C, H, W, stride, ho, wo);
        k_packw_offmask<<<(32 * C * 9 + 255) / 256, 256, 0, stream>>>(
            p[2], p[4], Bbuf2, C);
        k_gemm_offmask<<<M / 64, 256, 0, stream>>>(
            Abuf, Bbuf2, p[3], p[5], offb, modb, M, K, ho, wo);

        // Stage 2: deformable gather + main WMMA GEMM.
        k_im2col_dcn<<<(tIm + 255) / 256, 256, 0, stream>>>(
            xin, offb, modb, Abuf, B, C, H, W, stride, ho, wo);
        k_packw<<<(O * C * 9 + 255) / 256, 256, 0, stream>>>(p[0], Bbuf, C, O);
        k_gemm_dcn<<<M / 64, 256, 0, stream>>>(
            Abuf, Bbuf, p[1], q[0], q[1], q[2], q[3],
            addsrc, yout, M, K, O, ho, wo, relu);
    };

    run_dconv(x,  64, 64, 1, dc[0], bn[0], nullptr, y1, 1);        // y1 = relu(bn1(dc1(x)))
    run_dconv(y1, 64, 64, 2, dc[1], bn[1], nullptr, y2, 1);        // y2 = relu(bn2(dc2(y1)))
    run_dconv(y2, 32, 32, 1, dc[2], bn[2], nullptr, y3, 0);        // y3 = bn3(dc3(y2))
    run_dconv(x,  64, 64, 2, dc[3], bn[3], y3, (float*)d_out, 1);  // out = relu(bnid(dcid(x)) + y3)
}
